// MultiHeadAttention_19464791785581
// MI455X (gfx1250) — matrix operand order repaired and checked by reading, not compiled, not run on hardware
//
#include <hip/hip_runtime.h>

// ---------------------------------------------------------------------------
// MHA forward, MI455X (gfx1250): bf16 WMMA everywhere, fp32 accumulate,
// flash-style online softmax with async-to-LDS K/V pipeline (inline asm),
// output reduced over S then tiny GEMV.
// ---------------------------------------------------------------------------

typedef __bf16 bf16;
typedef __attribute__((ext_vector_type(16))) __bf16 v16bf;
typedef __attribute__((ext_vector_type(8)))  __bf16 v8bf;
typedef __attribute__((ext_vector_type(2)))  __bf16 v2bf;
typedef __attribute__((ext_vector_type(16))) float  v16f;
typedef __attribute__((ext_vector_type(8)))  float  v8f;
typedef __attribute__((ext_vector_type(4)))  float  v4f;
typedef __attribute__((ext_vector_type(2)))  float  v2f;

#define BATCH 2
#define SEQ   2048
#define DMODEL 1024
#define HEADS 16
#define DK    64

// ---- bf16 conversion: let the backend pick the native cvt lowering -------
#if __has_builtin(__builtin_amdgcn_cvt_pk_bf16_f32)
__device__ __forceinline__ v2bf f2bf2(float a, float b) {
    return __builtin_amdgcn_cvt_pk_bf16_f32(a, b);
}
__device__ __forceinline__ bf16 f2bf(float f) {
    return __builtin_amdgcn_cvt_pk_bf16_f32(f, 0.0f)[0];
}
#else
__device__ __forceinline__ bf16 f2bf(float f) { return (bf16)f; }
__device__ __forceinline__ v2bf f2bf2(float a, float b) {
    v2f x = {a, b};
    return __builtin_convertvector(x, v2bf);
}
#endif

__device__ __forceinline__ v8f wmma_bf16(v16bf a, v16bf b, v8f c) {
    return __builtin_amdgcn_wmma_f32_16x16x32_bf16(
        /*neg_a=*/false, a, /*neg_b=*/false, b,
        /*c_mod=*/(short)0, c, /*reuse_a=*/false, /*reuse_b=*/false);
}

// A-fragment (16x32, contiguous K layout) from a bf16 row; c0 = (lane>=16)*16
__device__ __forceinline__ v16bf load_a_bf16(const bf16* p, int c0) {
    v8bf lo = *(const v8bf*)(p + c0);
    v8bf hi = *(const v8bf*)(p + c0 + 8);
    v16bf a;
#pragma unroll
    for (int i = 0; i < 8; ++i) { a[i] = lo[i]; a[8 + i] = hi[i]; }
    return a;
}

// A-fragment from fp32 row (vector fptrunc -> packed cvt when available)
__device__ __forceinline__ v16bf load_a_f32(const float* p, int c0) {
    const float* q = p + c0;
    v4f f0 = *(const v4f*)(q);
    v4f f1 = *(const v4f*)(q + 4);
    v4f f2 = *(const v4f*)(q + 8);
    v4f f3 = *(const v4f*)(q + 12);
    v16f x;
#pragma unroll
    for (int i = 0; i < 4; ++i) {
        x[i] = f0[i]; x[4 + i] = f1[i]; x[8 + i] = f2[i]; x[12 + i] = f3[i];
    }
    return __builtin_convertvector(x, v16bf);
}

// ---- async global -> LDS staging (CDNA5 ASYNCcnt path), via inline asm ----
// ISA: GLOBAL_LOAD_ASYNC_TO_LDS_B128  VDST = per-lane LDS byte addr (VGPR),
//      VADDR = 64-bit global address, SADDR = off (GV mode).
__device__ __forceinline__ void async_cp16(const bf16* g, bf16* l) {
    unsigned lds_off = (unsigned)(size_t)l;   // LDS flat addr: offset in [31:0]
    asm volatile("global_load_async_to_lds_b128 %0, %1, off"
                 :: "v"(lds_off), "v"(g)
                 : "memory");
}

__device__ __forceinline__ void wait_async_batch(bool one_batch_in_flight) {
#if __has_builtin(__builtin_amdgcn_s_wait_asynccnt)
    if (one_batch_in_flight) __builtin_amdgcn_s_wait_asynccnt(4);
    else                     __builtin_amdgcn_s_wait_asynccnt(0);
#else
    if (one_batch_in_flight) asm volatile("s_wait_asynccnt 0x4" ::: "memory");
    else                     asm volatile("s_wait_asynccnt 0x0" ::: "memory");
#endif
}

// stage one 32-key chunk: K tile [32][64], V tile [64][32]; 4 x b128 / thread
__device__ __forceinline__ void stage_kv(const bf16* khb, const bf16* vtb, int kb,
                                         bf16* kbuf, bf16* vbuf, int t) {
#pragma unroll
    for (int r = 0; r < 2; ++r) {
        int seg = t + r * 128;                 // 0..255 (16B segments)
        int key = seg >> 3, off = (seg & 7) * 8;
        async_cp16(khb + (size_t)(kb + key) * DK + off, kbuf + key * DK + off);
    }
#pragma unroll
    for (int r = 0; r < 2; ++r) {
        int seg = t + r * 128;
        int d = seg >> 2, off = (seg & 3) * 8;
        async_cp16(vtb + (size_t)d * SEQ + kb + off, vbuf + d * 32 + off);
    }
}

// ---------------------------------------------------------------------------
// Kernel 1: W [in,out] fp32 -> Wt [out,in] bf16 (so B-fragments are contiguous)
// ---------------------------------------------------------------------------
__global__ void wconv_kernel(const float* __restrict__ W, bf16* __restrict__ Wt) {
    int idx = blockIdx.x * blockDim.x + threadIdx.x;       // over D*D
    int n = idx & (DMODEL - 1);
    int k = idx >> 10;
    Wt[(size_t)n * DMODEL + k] = f2bf(W[idx]);
}

// ---------------------------------------------------------------------------
// Kernel 2: X[B*S, D] @ W[D, D] + bias -> bf16 heads.
// mode 0/1: out[b,h,s,dk] row-major   mode 2: out[b,h,dk,s] (transposed V)
// Wave computes 32(M) x 64(N): two A-strips share four B-fragments.
// ---------------------------------------------------------------------------
__global__ __launch_bounds__(128) void proj_kernel(
    const float* __restrict__ X, const bf16* __restrict__ Wt,
    const float* __restrict__ bias, bf16* __restrict__ out,
    int mode, float scale)
{
    const int lane = threadIdx.x & 31;
    const int wave = threadIdx.x >> 5;
    const int nbase = blockIdx.x * 64;
    const int mbase = blockIdx.y * 128 + wave * 32;
    const int col  = lane & 15;
    const int half = lane >> 4;
    const int c0A = half * 16;
    const int c0B = half * 16;

    const float* xrow0 = X + (size_t)(mbase + col) * DMODEL;
    const float* xrow1 = X + (size_t)(mbase + 16 + col) * DMODEL;
    v8f acc[8] = {v8f{}, v8f{}, v8f{}, v8f{}, v8f{}, v8f{}, v8f{}, v8f{}};

    for (int kk = 0; kk < DMODEL; kk += 32) {
        if (kk + 32 < DMODEL)
            __builtin_prefetch(xrow0 + kk + 32, 0, 1);     // global_prefetch
        v16bf a0 = load_a_f32(xrow0 + kk, c0A);
        v16bf a1 = load_a_f32(xrow1 + kk, c0A);
#pragma unroll
        for (int nc = 0; nc < 4; ++nc) {
            int n = nbase + nc * 16 + col;
            v16bf b = *(const v16bf*)(Wt + (size_t)n * DMODEL + kk + c0B);
            acc[nc]     = wmma_bf16(a0, b, acc[nc]);
            acc[4 + nc] = wmma_bf16(a1, b, acc[4 + nc]);
        }
    }

#pragma unroll
    for (int strip = 0; strip < 2; ++strip) {
        const int mb = mbase + strip * 16;
        const int bb = mb >> 11;                 // batch (strip never straddles)
#pragma unroll
        for (int nc = 0; nc < 4; ++nc) {
            int n  = nbase + nc * 16 + col;
            int h  = n >> 6;
            int dk = n & 63;
            float bs = bias[n];
            const v8f& a = acc[strip * 4 + nc];
            float w[8];
#pragma unroll
            for (int e = 0; e < 8; ++e) w[e] = (a[e] + bs) * scale;

            if (mode == 2) {
                // vhT[b,h,dk,s]: the 8 elements are consecutive s -> b128 store
                int s = (mb + half * 8) & (SEQ - 1);
                bf16* dst = out + (((size_t)bb * HEADS + h) * DK + dk) * SEQ + s;
                v8bf pk;
#pragma unroll
                for (int i = 0; i < 4; ++i) {
                    v2bf t = f2bf2(w[2 * i], w[2 * i + 1]);
                    pk[2 * i] = t[0]; pk[2 * i + 1] = t[1];
                }
                *(v8bf*)dst = pk;
            } else {
                // qh/kh[b,h,s,dk]: rows strided -> scalar b16 stores
#pragma unroll
                for (int e = 0; e < 8; ++e) {
                    int s = (mb + e + half * 8) & (SEQ - 1);
                    out[(((size_t)bb * HEADS + h) * SEQ + s) * DK + dk] = f2bf(w[e]);
                }
            }
        }
    }
}

// ---------------------------------------------------------------------------
// Kernel 3: flash attention. Block = 4 waves sharing one (b,h); K/V chunks
// staged into double-buffered LDS with async loads; each wave owns a
// 16-query tile; row-gated column sums atomically added into y[b, h*64+dk].
// ---------------------------------------------------------------------------
__global__ __launch_bounds__(128) void attn_kernel(
    const bf16* __restrict__ qh, const bf16* __restrict__ kh,
    const bf16* __restrict__ vhT, const int* __restrict__ mask,
    float* __restrict__ y)
{
    __shared__ bf16 skbuf[2 * 32 * DK];     // K tiles  [2][32][64]  8 KB
    __shared__ bf16 svbuf[2 * DK * 32];     // V tiles  [2][64][32]  8 KB
    __shared__ bf16 plds[4 * 16 * 32];      // per-wave p tile       4 KB

    const int lane = threadIdx.x & 31;
    const int wave = threadIdx.x >> 5;
    const int gw   = blockIdx.x * 4 + wave;     // 0..4095
    const int qt   = gw & 127;                  // query tile within (b,h)
    const int bh   = gw >> 7;                   // same for all 4 waves
    const int b    = bh >> 4;
    const int qbase = qt * 16;

    const int col  = lane & 15;
    const int half = lane >> 4;
    const int c0A = half * 16;
    const int c0B = half * 16;

    // q A-fragments (Dk=64 -> two K=32 chunks), loaded once
    const bf16* qrow = qh + ((size_t)bh * SEQ + (qbase + col)) * DK;
    v16bf qa0 = load_a_bf16(qrow, c0A);
    v16bf qa1 = load_a_bf16(qrow + 32, c0A);

    const bf16* khb = kh  + (size_t)bh * SEQ * DK;
    const bf16* vtb = vhT + (size_t)bh * DK * SEQ;

    v8f o0{}, o1{}, o2{}, o3{};
    float mrun[8], lrun[8];
#pragma unroll
    for (int e = 0; e < 8; ++e) { mrun[e] = -3.0e38f; lrun[e] = 0.0f; }

    bf16* pl = plds + wave * 512;

    // prologue: stage chunk 0
    stage_kv(khb, vtb, 0, skbuf, svbuf, threadIdx.x);

    for (int kb = 0, it = 0; kb < SEQ; kb += 32, ++it) {
        const int cur = it & 1;
        const bool more = (kb + 32 < SEQ);
        if (more)   // overlap next chunk's DMA with this chunk's WMMAs
            stage_kv(khb, vtb, kb + 32,
                     skbuf + (cur ^ 1) * 32 * DK,
                     svbuf + (cur ^ 1) * DK * 32, threadIdx.x);
        wait_async_batch(more);     // s_wait_asynccnt (my chunk landed)
        __syncthreads();            // everyone's pieces visible

        const bf16* kl = skbuf + cur * 32 * DK;
        const bf16* vl = svbuf + cur * DK * 32;

        // scores: s = q (16x64) x kh^T (64x32), two 16x16 D-tiles, from LDS
        const bf16* k0r = kl + col * DK;
        const bf16* k1r = kl + (16 + col) * DK;
        v16bf kb0a = *(const v16bf*)(k0r + c0B);
        v16bf kb0b = *(const v16bf*)(k0r + 32 + c0B);
        v16bf kb1a = *(const v16bf*)(k1r + c0B);
        v16bf kb1b = *(const v16bf*)(k1r + 32 + c0B);
        v8f s0{}, s1{};
        s0 = wmma_bf16(qa0, kb0a, s0);
        s0 = wmma_bf16(qa1, kb0b, s0);
        s1 = wmma_bf16(qa0, kb1a, s1);
        s1 = wmma_bf16(qa1, kb1b, s1);

        // online softmax (per-row stats; rows live per C-layout element)
        float corr[8];
#pragma unroll
        for (int e = 0; e < 8; ++e) {
            float mx = fmaxf(s0[e], s1[e]);
            mx = fmaxf(mx, __shfl_xor(mx, 1));
            mx = fmaxf(mx, __shfl_xor(mx, 2));
            mx = fmaxf(mx, __shfl_xor(mx, 4));
            mx = fmaxf(mx, __shfl_xor(mx, 8));
            float mnew = fmaxf(mrun[e], mx);
            float c  = __expf(mrun[e] - mnew);
            float p0 = __expf(s0[e] - mnew);
            float p1 = __expf(s1[e] - mnew);
            float rs = p0 + p1;
            rs += __shfl_xor(rs, 1);
            rs += __shfl_xor(rs, 2);
            rs += __shfl_xor(rs, 4);
            rs += __shfl_xor(rs, 8);
            lrun[e] = lrun[e] * c + rs;
            mrun[e] = mnew;
            corr[e] = c;
            s0[e] = p0; s1[e] = p1;
        }
#pragma unroll
        for (int e = 0; e < 8; ++e) {
            o0[e] *= corr[e]; o1[e] *= corr[e];
            o2[e] *= corr[e]; o3[e] *= corr[e];
        }

        // relayout p: C-layout -> LDS 16x32 bf16 -> A-fragment (wave-private)
#pragma unroll
        for (int e = 0; e < 8; ++e) {
            int r = e + half * 8;
            pl[r * 32 + col]      = f2bf(s0[e]);
            pl[r * 32 + 16 + col] = f2bf(s1[e]);
        }
        v16bf pa = load_a_bf16(pl + col * 32, c0A);

        // o (16x64) += p (16x32) x vh (32x64) ; B-frag rows from LDS V tile
#pragma unroll
        for (int nc = 0; nc < 4; ++nc) {
            int d = nc * 16 + col;
            v16bf vb = *(const v16bf*)(vl + d * 32 + c0B);
            v8f* o = (nc == 0) ? &o0 : (nc == 1) ? &o1 : (nc == 2) ? &o2 : &o3;
            *o = wmma_bf16(pa, vb, *o);
        }
        __syncthreads();            // done reading this buffer
    }

    // finalize: 1/rowsum, query-row mask gate, then column sums -> y
    float inv[8];
#pragma unroll
    for (int e = 0; e < 8; ++e) {
        int r = qbase + e + half * 8;
        int gate = mask[b * SEQ + r];
        inv[e] = (gate != 0) ? (1.0f / lrun[e]) : 0.0f;
    }
    float* yb = y + b * DMODEL + (bh & 15) * 64;
#pragma unroll
    for (int nc = 0; nc < 4; ++nc) {
        const v8f& o = (nc == 0) ? o0 : (nc == 1) ? o1 : (nc == 2) ? o2 : o3;
        float cs = 0.0f;
#pragma unroll
        for (int e = 0; e < 8; ++e) cs += o[e] * inv[e];
        atomicAdd(&yb[nc * 16 + col], cs);     // global_atomic_add_f32
    }
}

// ---------------------------------------------------------------------------
// Kernel 4: per-batch count of unmasked rows (bias multiplier)
// ---------------------------------------------------------------------------
__global__ void maskcnt_kernel(const int* __restrict__ mask, int* __restrict__ cnt) {
    __shared__ int sm[256];
    int b = blockIdx.x, t = threadIdx.x;
    int s = 0;
    for (int i = t; i < SEQ; i += 256) s += (mask[b * SEQ + i] != 0) ? 1 : 0;
    sm[t] = s;
    __syncthreads();
    for (int off = 128; off > 0; off >>= 1) {
        if (t < off) sm[t] += sm[t + off];
        __syncthreads();
    }
    if (t == 0) cnt[b] = sm[0];
}

// ---------------------------------------------------------------------------
// Kernel 5: out[b,n] = y[b,:] . Wo[:,n] + cnt[b]*bo[n]   (tiny GEMV, fp32)
// ---------------------------------------------------------------------------
__global__ __launch_bounds__(256) void outproj_kernel(
    const float* __restrict__ y, const float* __restrict__ Wo,
    const float* __restrict__ bo, const int* __restrict__ cnt,
    float* __restrict__ out)
{
    int id = blockIdx.x * 256 + threadIdx.x;   // 0..B*D-1
    int b = id >> 10, n = id & (DMODEL - 1);
    const float* yb = y + b * DMODEL;
    float acc = 0.0f;
    for (int k = 0; k < DMODEL; ++k)
        acc += yb[k] * Wo[(size_t)k * DMODEL + n];
    out[id] = acc + (float)cnt[b] * bo[n];
}

// ---------------------------------------------------------------------------
extern "C" void kernel_launch(void* const* d_in, const int* in_sizes, int n_in,
                              void* d_out, int out_size, void* d_ws, size_t ws_size,
                              hipStream_t stream) {
    (void)in_sizes; (void)n_in; (void)out_size; (void)ws_size;

    const float* q    = (const float*)d_in[0];
    const float* k    = (const float*)d_in[1];
    const float* v    = (const float*)d_in[2];
    const int*   mask = (const int*)  d_in[3];
    const float* Wq   = (const float*)d_in[4];
    const float* bq   = (const float*)d_in[5];
    const float* Wk   = (const float*)d_in[6];
    const float* bk   = (const float*)d_in[7];
    const float* Wv   = (const float*)d_in[8];
    const float* bv   = (const float*)d_in[9];
    const float* Wo   = (const float*)d_in[10];
    const float* bo   = (const float*)d_in[11];
    float* out = (float*)d_out;

    // workspace layout
    char* w = (char*)d_ws;
    const size_t WT_BYTES  = (size_t)DMODEL * DMODEL * sizeof(bf16);       // 2 MB
    const size_t ACT_BYTES = (size_t)BATCH * SEQ * DMODEL * sizeof(bf16);  // 8 MB
    bf16* Wtq = (bf16*)(w);
    bf16* Wtk = (bf16*)(w + WT_BYTES);
    bf16* Wtv = (bf16*)(w + 2 * WT_BYTES);
    bf16* qh  = (bf16*)(w + 3 * WT_BYTES);
    bf16* kh  = (bf16*)(w + 3 * WT_BYTES + ACT_BYTES);
    bf16* vhT = (bf16*)(w + 3 * WT_BYTES + 2 * ACT_BYTES);
    float* y  = (float*)(w + 3 * WT_BYTES + 3 * ACT_BYTES);
    int*   cnt = (int*)(w + 3 * WT_BYTES + 3 * ACT_BYTES + 8192);

    (void)hipMemsetAsync(y, 0, (size_t)BATCH * DMODEL * sizeof(float), stream);

    // 1) weight transpose + bf16 convert
    const int WBLK = (DMODEL * DMODEL) / 256;
    wconv_kernel<<<WBLK, 256, 0, stream>>>(Wq, Wtq);
    wconv_kernel<<<WBLK, 256, 0, stream>>>(Wk, Wtk);
    wconv_kernel<<<WBLK, 256, 0, stream>>>(Wv, Wtv);

    // 2) QKV projections (1/sqrt(Dk)=0.125 folded into qh)
    dim3 pg(DMODEL / 64, (BATCH * SEQ) / 128);
    proj_kernel<<<pg, 128, 0, stream>>>(q, Wtq, bq, qh,  0, 0.125f);
    proj_kernel<<<pg, 128, 0, stream>>>(k, Wtk, bk, kh,  0, 1.0f);
    proj_kernel<<<pg, 128, 0, stream>>>(v, Wtv, bv, vhT, 2, 1.0f);

    // 3) attention + reduction over S into y
    attn_kernel<<<(BATCH * HEADS * (SEQ / 16)) / 4, 128, 0, stream>>>(
        qh, kh, vhT, mask, y);

    // 4) mask counts, 5) final projection
    maskcnt_kernel<<<BATCH, 256, 0, stream>>>(mask, cnt);
    outproj_kernel<<<(BATCH * DMODEL) / 256, 256, 0, stream>>>(y, Wo, bo, cnt, out);
}
